// predictor_67645734912742
// MI455X (gfx1250) — matrix operand order repaired and checked by reading, not compiled, not run on hardware
//
#include <hip/hip_runtime.h>

// ---------------- CDNA5 (gfx1250) fused sparse-GCN kernel, v3 ----------------
// One workgroup (256 threads = 8 wave32) per batch sample.
//   1. x[b] -> zero-padded 224x224 f32 LDS tile via GLOBAL_LOAD_ASYNC_TO_LDS_B128
//      (no VGPR round-trip; completion via s_wait_asynccnt).
//   2. Exact top-k threshold via 4x8-bit radix select over float bits in LDS.
//   3. deg/dinv (col sums) and u_i = col-sums of An^T (collapses layer-2 mean).
//   4. bf16 WMMA: XW1 = X@W1 ; H1 = relu(An^T @ XW1 + b1)
//      B operands stored TRANSPOSED [n][k]: one 32B LDS read per B fragment,
//      one 16B LDS store per D-tile writeback.
//   5. out = ((u^T H1) @ W2)/200 + b2  -- two GEMVs, no third GEMM needed.

typedef __attribute__((ext_vector_type(16))) __bf16 v16bf;
typedef __attribute__((ext_vector_type(8)))  __bf16 bf16x8;
typedef __attribute__((ext_vector_type(8)))  float  v8f;

#define NN   200          // nodes
#define HH   128          // hidden
#define NT   13           // row tiles (13*16 = 208)
#define KT1  7            // k-steps of 32 (224 >= 200)
#define MP   224          // padded major dim / K stride
#define KEEP 8000         // int(200*200*0.2)

__global__ __launch_bounds__(256, 1)
void gcn_sparse_fused(const float* __restrict__ x,
                      const float* __restrict__ W1,
                      const float* __restrict__ b1,
                      const float* __restrict__ W2,
                      const float* __restrict__ b2,
                      float* __restrict__ out)
{
  __shared__ __align__(16) float  xs[MP * MP];     // 200704 B, zero-padded x
  __shared__ __align__(32) __bf16 xwT[HH * MP];    //  57344 B, XW1/H1 transposed [h][m]
  __shared__ __align__(32) __bf16 wstT[HH * MP];   //  57344 B, W1 transposed [h][k]
  __shared__ float    dinv[MP];
  __shared__ float    uvec[MP];
  __shared__ float    vpart[256];
  __shared__ float    vvec[HH];
  __shared__ unsigned hist[256];
  __shared__ unsigned sel_prefix;
  __shared__ unsigned sel_kk;
  __shared__ float    thr_s;

  const int tid  = threadIdx.x;
  const int lane = tid & 31;
  const int wave = tid >> 5;            // 0..7 -> owns column strip n0 = wave*16
  const int nloc = lane & 15;
  const int hi   = (lane >= 16);        // lane half selects K sub-range
  const int moff = hi ? 8 : 0;
  const int bidx = blockIdx.x;

  // ---- zero padded LDS regions (DScnt-tracked; fenced by the barrier) ----
  {
    const float4 z4 = make_float4(0.f, 0.f, 0.f, 0.f);
    for (int i = tid; i < (MP * MP) / 4; i += 256) ((float4*)xs)[i] = z4;
    for (int i = tid; i < (HH * MP) / 2; i += 256) ((unsigned*)xwT)[i] = 0u;
    if (tid == 0) { sel_prefix = 0u; sel_kk = KEEP; }
  }
  __syncthreads();

  // ---- stage x rows cache->LDS asynchronously (ASYNCcnt path, no VGPR bounce)
  {
    const float4* xg = (const float4*)(x + (size_t)bidx * (NN * NN));
    const unsigned lds0 = (unsigned)(size_t)(void*)xs;
    for (int i = tid; i < NN * (NN / 4); i += 256) {
      int r = i / (NN / 4), c4 = i - r * (NN / 4);
      unsigned laddr = lds0 + (unsigned)((r * MP + c4 * 4) * 4);
      const float4* gp = xg + i;
      asm volatile("global_load_async_to_lds_b128 %0, %1, off"
                   :: "v"(laddr), "v"(gp) : "memory");
    }
    // W1 transposed bf16 (normal path, overlaps with in-flight async loads)
    for (int i = tid; i < HH * MP; i += 256) {
      int k = i >> 7, h = i & (HH - 1);             // global-coalesced over h
      wstT[h * MP + k] = (__bf16)((k < NN) ? W1[k * HH + h] : 0.0f);
    }
    asm volatile("s_wait_asynccnt 0" ::: "memory");
  }
  __syncthreads();

  // ---- exact top-k threshold: 4-pass radix select on float bits ----
  for (int p = 0; p < 4; ++p) {
    const int shift = 24 - 8 * p;
    hist[tid] = 0u;
    __syncthreads();
    const unsigned pref = sel_prefix;
    for (int r = wave; r < NN; r += 8) {
      for (int c = lane; c < NN; c += 32) {
        unsigned key = __float_as_uint(xs[r * MP + c]);   // x >= 0 -> monotone
        unsigned hb  = (p == 0) ? 0u : (key >> (shift + 8));
        if (hb == pref) atomicAdd(&hist[(key >> shift) & 255u], 1u);
      }
    }
    __syncthreads();
    if (tid == 0) {
      unsigned kk = sel_kk, cum = 0; int d = 255;
      for (; d > 0; --d) { unsigned c = hist[d]; if (cum + c >= kk) break; cum += c; }
      sel_prefix = (sel_prefix << 8) | (unsigned)d;
      sel_kk = kk - cum;
    }
    __syncthreads();
  }
  if (tid == 0) thr_s = __uint_as_float(sel_prefix);
  __syncthreads();
  const float thr = thr_s;              // keep x >= thr (exactly KEEP elements)

  // ---- weighted in-degree (col sums) -> dinv (zero-extended to MP) ----
  if (tid < MP) {
    float dv = 0.0f;
    if (tid < NN) {
      float deg = 1.0f;                              // self loop
      for (int i = 0; i < NN; ++i) {
        float v = xs[i * MP + tid];
        deg += (v >= thr) ? v : 0.0f;
      }
      dv = __frsqrt_rn(deg);
    }
    dinv[tid] = dv;
  }
  __syncthreads();
  // ---- u_i = dinv_i * (dinv_i + sum_j dinv_j * A_ij) ----
  if (tid < MP) {
    float uv = 0.0f;
    if (tid < NN) {
      float s = 0.0f;
      for (int j = 0; j < NN; ++j) {
        float v = xs[tid * MP + j];
        s += (v >= thr) ? dinv[j] * v : 0.0f;
      }
      uv = dinv[tid] * (dinv[tid] + s);
    }
    uvec[tid] = uv;
  }
  __syncthreads();

  const int n0    = wave * 16;
  const int ncol  = n0 + nloc;                       // this lane's output column
  const int kselA = hi ? 16 : 0;                     // A-frag K sub-offset
  const int kselB = hi ? 16 : 0;                     // B-frag K sub-offset

  // ================= GEMM a : XW1 = X @ W1 -> xwT (bf16) =================
  for (int mt = 0; mt < NT; ++mt) {
    v8f acc = {};
    const float* rowp = &xs[(mt * 16 + nloc) * MP];  // this lane's A row
    for (int kt = 0; kt < KT1; ++kt) {
      const float* rp = rowp + kt * 32 + kselA;
      const float4 va = *(const float4*)(rp + 0);    // K = off+0..3
      const float4 vb = *(const float4*)(rp + 4);    // K = off+4..7
      const float4 vc = *(const float4*)(rp + 8);    // K = off+8..11
      const float4 vd = *(const float4*)(rp + 12);   // K = off+12..15
      v16bf a;
      a[0]=(__bf16)va.x; a[1]=(__bf16)va.y; a[2]=(__bf16)va.z; a[3]=(__bf16)va.w;
      a[4]=(__bf16)vb.x; a[5]=(__bf16)vb.y; a[6]=(__bf16)vb.z; a[7]=(__bf16)vb.w;
      a[8]=(__bf16)vc.x; a[9]=(__bf16)vc.y; a[10]=(__bf16)vc.z; a[11]=(__bf16)vc.w;
      a[12]=(__bf16)vd.x; a[13]=(__bf16)vd.y; a[14]=(__bf16)vd.z; a[15]=(__bf16)vd.w;
      const v16bf bfr = *(const v16bf*)&wstT[ncol * MP + kt * 32 + kselB];
      acc = __builtin_amdgcn_wmma_f32_16x16x32_bf16(false, a, false, bfr,
                                                    (short)0, acc, false, false);
    }
    bf16x8 pk;
    #pragma unroll
    for (int r = 0; r < 8; ++r) pk[r] = (__bf16)acc[r];
    *(bf16x8*)&xwT[ncol * MP + mt * 16 + moff] = pk; // one 16B store
  }
  __syncthreads();

  // ===== GEMM b : T1 = (An^T) @ XW1, An^T built on the fly from xs =====
  v8f accs[NT];
  #pragma unroll
  for (int mt = 0; mt < NT; ++mt) {
    v8f acc = {};
    const int jj = mt * 16 + nloc;                   // output row j (< 208)
    const float dj = dinv[jj];                       // 0 for jj >= 200
    for (int kt = 0; kt < KT1; ++kt) {
      v16bf a;
      #pragma unroll
      for (int e = 0; e < 16; ++e) {
        const int ia = kt * 32 + e + kselA;  // source node
        const float xv = xs[ia * MP + jj];           // transpose read (padded 0)
        float ah = (xv >= thr) ? xv : 0.0f;          // sparsified edge
        ah += (ia == jj) ? 1.0f : 0.0f;              // self loop
        a[e] = (__bf16)(dj * dinv[ia] * ah);         // symmetric normalization
      }
      const v16bf bfr = *(const v16bf*)&xwT[ncol * MP + kt * 32 + kselB];
      acc = __builtin_amdgcn_wmma_f32_16x16x32_bf16(false, a, false, bfr,
                                                    (short)0, acc, false, false);
    }
    accs[mt] = acc;                                  // keep in VGPRs
  }
  __syncthreads();                                   // all xwT (XW1) reads done

  // ---- H1 = relu(T1 + b1) -> xwT in place ----
  {
    const float b1v = b1[ncol];
    #pragma unroll
    for (int mt = 0; mt < NT; ++mt) {
      bf16x8 pk;
      #pragma unroll
      for (int r = 0; r < 8; ++r) {
        float v = accs[mt][r] + b1v;
        pk[r] = (__bf16)((v > 0.0f) ? v : 0.0f);
      }
      *(bf16x8*)&xwT[ncol * MP + mt * 16 + moff] = pk;
    }
  }
  __syncthreads();

  // ---- v = u^T @ H1 : two threads per column, halves of the m range ----
  {
    const int h    = tid & (HH - 1);
    const int half = tid >> 7;                       // 0 or 1
    const int m0   = half * 104, m1 = m0 + 104;      // covers 0..207
    float s = 0.0f;
    for (int m = m0; m < m1; ++m)                    // u[m]=0 for m>=200
      s += uvec[m] * (float)xwT[h * MP + m];
    vpart[tid] = s;
  }
  __syncthreads();
  if (tid < HH) vvec[tid] = vpart[tid] + vpart[tid + HH];
  __syncthreads();

  // ---- out = (v @ W2)/200 + b2 ----
  if (tid < HH) {
    float s = 0.0f;
    for (int h1 = 0; h1 < HH; ++h1)
      s += vvec[h1] * W2[h1 * HH + tid];             // coalesced, L2-resident
    out[(size_t)bidx * HH + tid] = s * (1.0f / (float)NN) + b2[tid];
  }
}

extern "C" void kernel_launch(void* const* d_in, const int* in_sizes, int n_in,
                              void* d_out, int out_size, void* d_ws, size_t ws_size,
                              hipStream_t stream) {
  (void)in_sizes; (void)n_in; (void)d_ws; (void)ws_size; (void)out_size;
  const float* x  = (const float*)d_in[0];
  // d_in[1] = adj : unused — the reference overwrites it with sparse(x, 0.2)
  const float* W1 = (const float*)d_in[2];
  const float* b1 = (const float*)d_in[3];
  const float* W2 = (const float*)d_in[4];
  const float* b2 = (const float*)d_in[5];
  float* out = (float*)d_out;

  gcn_sparse_fused<<<1024, 256, 0, stream>>>(x, W1, b1, W2, b2, out);
}
